// TransformerEncoderLayer_29016799052500
// MI455X (gfx1250) — matrix operand order repaired and checked by reading, not compiled, not run on hardware
//
#include <hip/hip_runtime.h>

#define Bn 4
#define Ln 256
#define Dn 512
#define Hn 8
#define HDn 64
#define DFFn 2048
#define BLn (Bn * Ln)

typedef __attribute__((ext_vector_type(16))) __bf16 v16bf;
typedef __attribute__((ext_vector_type(8)))  float v8f;

struct alignas(16) U128 { unsigned int x, y, z, w; };
union FragU { U128 q[2]; v16bf v; };

__device__ __forceinline__ unsigned short f2bf(float f) {
  unsigned int u = __float_as_uint(f);
  u += 0x7FFFu + ((u >> 16) & 1u);
  return (unsigned short)(u >> 16);
}

// Async DMA: global -> LDS, 16 bytes per lane, tracked by ASYNCcnt (CDNA5 §10.7/§15.18).
__device__ __forceinline__ void async_b128(unsigned int lds_byte_off, const void* gaddr) {
  asm volatile("global_load_async_to_lds_b128 %0, %1, off"
               :: "v"(lds_byte_off), "v"(gaddr) : "memory");
}
__device__ __forceinline__ void wait_async0() {
  asm volatile("s_wait_asynccnt 0x0" ::: "memory");
}

// ---------------------------------------------------------------- convert
__global__ __launch_bounds__(256) void convert_bf16_kernel(
    const float* __restrict__ in, unsigned short* __restrict__ out, int n) {
  int i = blockIdx.x * 256 + threadIdx.x;
  if (i < n) out[i] = f2bf(in[i]);
}

// ---------------------------------------------------------------- bf16 WMMA GEMM: C[m,n] = sum_k A[m,k] * Bm[n,k]  (TN)
// 256 threads = 8 waves. Block tile 64 x (64*WNT); wave tile 32 x (16*WNT).
// Double-buffered LDS fed by global_load_async_to_lds_b128.
// Requires M % 64 == 0, N % (64*WNT) == 0, K % 32 == 0 (all call sites conform).
template <int WNT>
__global__ __launch_bounds__(256) void gemm_bf16_wmma_kernel(
    const unsigned short* __restrict__ A, long long sA, int lda,
    const unsigned short* __restrict__ Bm, long long sB, int ldb,
    float* __restrict__ Cf, unsigned short* __restrict__ Cb,
    int zdiv, long long csOuter, long long csInner, int ldc,
    const float* __restrict__ bias, int relu, int M, int N, int K) {
  constexpr int ASTR = 40;                 // padded row stride (bf16): 80B, 16B aligned
  constexpr int A_TILE = 64 * ASTR;        // ushorts per A buffer
  constexpr int B_TILE = 64 * WNT * ASTR;  // ushorts per B buffer
  __shared__ unsigned short smem[2 * A_TILE + 2 * B_TILE];

  const int z = blockIdx.z;
  const unsigned short* Ab = A + (size_t)z * sA;
  const unsigned short* Bb = Bm + (size_t)z * sB;
  const long long offC = (long long)(z / zdiv) * csOuter + (long long)(z % zdiv) * csInner;

  const int m_base = blockIdx.y * 64;
  const int n_base = blockIdx.x * (64 * WNT);
  const int t    = threadIdx.x;
  const int lane = t & 31;
  const int wave = t >> 5;
  const int wm   = wave >> 2;   // 0..1 : 32 rows of M
  const int wn   = wave & 3;    // 0..3 : 16*WNT cols of N
  const int half = lane >> 4;
  const int lrow = lane & 15;
  const int sr = t >> 2;        // staging row 0..63
  const int sc = (t & 3) * 8;   // staging col {0,8,16,24}

  // Issue one K-step's DMA into buffer `buf` (1 + WNT b128 copies per thread).
  auto issue = [&](int buf, int k0) {
    const unsigned int la = (unsigned int)(buf * A_TILE + sr * ASTR + sc) * 2u;
    async_b128(la, Ab + (size_t)(m_base + sr) * lda + k0 + sc);
#pragma unroll
    for (int i = 0; i < WNT; ++i) {
      const unsigned int lb =
          (unsigned int)(2 * A_TILE + buf * B_TILE + (sr + 64 * i) * ASTR + sc) * 2u;
      async_b128(lb, Bb + (size_t)(n_base + sr + 64 * i) * ldb + k0 + sc);
    }
  };

  v8f acc[2][WNT];
  const v8f z8 = {0.f, 0.f, 0.f, 0.f, 0.f, 0.f, 0.f, 0.f};
#pragma unroll
  for (int s = 0; s < 2; ++s)
#pragma unroll
    for (int j = 0; j < WNT; ++j) acc[s][j] = z8;

  issue(0, 0);
  wait_async0();
  __syncthreads();

  int buf = 0;
  for (int k0 = 0; k0 < K; k0 += 32) {
    if (k0 + 32 < K) issue(buf ^ 1, k0 + 32);   // DMA next stage, overlapped with WMMA

    const unsigned short* sa_ = &smem[buf * A_TILE];
    const unsigned short* sb_ = &smem[2 * A_TILE + buf * B_TILE];

    // A fragments (ISA 7.12.2): lane holds row M=lrow; e<8 -> K=e+8*half ; e>=8 -> K=e+8+8*half
    FragU fa[2];
#pragma unroll
    for (int s = 0; s < 2; ++s) {
      const int ar = wm * 32 + s * 16 + lrow;
      fa[s].q[0] = *(const U128*)&sa_[ar * ASTR + 8 * half];
      fa[s].q[1] = *(const U128*)&sa_[ar * ASTR + 16 + 8 * half];
    }
    // B fragments: lane holds col N; e<8 -> K=e+8*half ; e>=8 -> K=e+8+8*half
    FragU fb[WNT];
#pragma unroll
    for (int j = 0; j < WNT; ++j) {
      const int br = wn * (16 * WNT) + 16 * j + lrow;
      fb[j].q[0] = *(const U128*)&sb_[br * ASTR + 8 * half];
      fb[j].q[1] = *(const U128*)&sb_[br * ASTR + 16 + 8 * half];
    }
#pragma unroll
    for (int s = 0; s < 2; ++s)
#pragma unroll
      for (int j = 0; j < WNT; ++j)
        acc[s][j] = __builtin_amdgcn_wmma_f32_16x16x32_bf16(
            false, fa[s].v, false, fb[j].v, (short)0, acc[s][j], false, false);

    wait_async0();     // next stage fully landed in LDS
    __syncthreads();   // all waves done reading current stage / writes visible
    buf ^= 1;
  }

#pragma unroll
  for (int j = 0; j < WNT; ++j) {
    const int col = n_base + wn * (16 * WNT) + 16 * j + lrow;
    const float bv = bias ? bias[col] : 0.0f;
#pragma unroll
    for (int s = 0; s < 2; ++s) {
#pragma unroll
      for (int e = 0; e < 8; ++e) {
        const int r = m_base + wm * 32 + s * 16 + (e + 8 * half);  // C: VGPR e -> M=e+8*half
        float v = acc[s][j][e] + bv;
        if (relu) v = fmaxf(v, 0.f);
        if (Cf) Cf[offC + (size_t)r * ldc + col] = v;
        if (Cb) Cb[offC + (size_t)r * ldc + col] = f2bf(v);
      }
    }
  }
}

// ---------------------------------------------------------------- qkv split / scale / bias_u,v / head layout
__global__ __launch_bounds__(256) void prep_qkv_kernel(
    const float* __restrict__ qkv, const float* __restrict__ bu, const float* __restrict__ bvv,
    unsigned short* __restrict__ quB, float* __restrict__ qvF,
    unsigned short* __restrict__ kB, unsigned short* __restrict__ vT) {
  const int bl = blockIdx.x;              // 0..1023
  const int b = bl >> 8, l = bl & 255;
  const float* row = qkv + (size_t)bl * (3 * Dn);
  for (int dd = threadIdx.x; dd < Dn; dd += 256) {
    const int h = dd >> 6, d = dd & 63;
    const float q  = row[dd] * 0.125f;          // 1/sqrt(64)
    const float kk = row[Dn + dd];
    const float vv = row[2 * Dn + dd];
    const size_t idx = (((size_t)(b * Hn + h)) * Ln + l) * HDn + d;
    quB[idx] = f2bf(q + bu[dd]);
    qvF[idx] = q + bvv[dd];
    kB[idx]  = f2bf(kk);
    vT[(((size_t)(b * Hn + h)) * HDn + d) * Ln + l] = f2bf(vv);
  }
}

// ---------------------------------------------------------------- b_d term: S[b,h,l,w] += dot(qv[b,h,l,:], pf[b,l,w,h*64:...])
__global__ __launch_bounds__(256) void bd_add_kernel(
    const float* __restrict__ qv, const float* __restrict__ pf, float* __restrict__ S) {
  const int l = blockIdx.x, b = blockIdx.y;
  __shared__ float sqv[Dn];
  const int t = threadIdx.x;
#pragma unroll
  for (int i = t; i < Dn; i += 256) {
    const int h = i >> 6, d = i & 63;
    sqv[i] = qv[(((size_t)(b * Hn + h)) * Ln + l) * HDn + d];
  }
  __syncthreads();

  const int h = t & 7;
  const int w0 = t >> 3;                  // 32 w slots
  const float4* q4 = (const float4*)&sqv[h * HDn];
  const float* pfbase = pf + (((size_t)b * Ln + l) * Ln) * Dn;
  for (int w = w0; w < Ln; w += 32) {
    const float4* p4 = (const float4*)(pfbase + (size_t)w * Dn + h * HDn);
    float acc = 0.f;
#pragma unroll
    for (int d4 = 0; d4 < 16; ++d4) {
      const float4 p = p4[d4];
      const float4 q = q4[d4];
      acc += p.x * q.x + p.y * q.y + p.z * q.z + p.w * q.w;
    }
    S[(((size_t)(b * Hn + h)) * Ln + l) * Ln + w] += acc;
  }
}

// ---------------------------------------------------------------- softmax over last dim (256), one wave per row -> bf16 probs
__global__ __launch_bounds__(256) void softmax_kernel(
    const float* __restrict__ S, unsigned short* __restrict__ P) {
  const int row = blockIdx.x * 8 + (threadIdx.x >> 5);
  const int lane = threadIdx.x & 31;
  const float* s = S + (size_t)row * Ln;
  float x[8];
#pragma unroll
  for (int j = 0; j < 8; ++j) x[j] = s[lane + 32 * j];
  float mx = x[0];
#pragma unroll
  for (int j = 1; j < 8; ++j) mx = fmaxf(mx, x[j]);
#pragma unroll
  for (int o = 16; o > 0; o >>= 1) mx = fmaxf(mx, __shfl_xor(mx, o, 32));
  float sum = 0.f;
#pragma unroll
  for (int j = 0; j < 8; ++j) { x[j] = __expf(x[j] - mx); sum += x[j]; }
#pragma unroll
  for (int o = 16; o > 0; o >>= 1) sum += __shfl_xor(sum, o, 32);
  const float inv = 1.0f / sum;
  unsigned short* p = P + (size_t)row * Ln;
#pragma unroll
  for (int j = 0; j < 8; ++j) p[lane + 32 * j] = f2bf(x[j] * inv);
}

// ---------------------------------------------------------------- y = LN(a + r) * g + be ; one wave per 512-wide row
__global__ __launch_bounds__(256) void add_ln_kernel(
    const float* __restrict__ a, const float* __restrict__ r,
    const float* __restrict__ g, const float* __restrict__ be,
    float* __restrict__ outF, unsigned short* __restrict__ outB) {
  const int row = blockIdx.x * 8 + (threadIdx.x >> 5);
  const int lane = threadIdx.x & 31;
  const float4* av = (const float4*)(a + (size_t)row * Dn);
  const float4* rv = (const float4*)(r + (size_t)row * Dn);
  float4 x[4];
  float sum = 0.f, sq = 0.f;
#pragma unroll
  for (int j = 0; j < 4; ++j) {
    const int c = lane + 32 * j;
    const float4 xa = av[c], xr = rv[c];
    float4 v;
    v.x = xa.x + xr.x; v.y = xa.y + xr.y; v.z = xa.z + xr.z; v.w = xa.w + xr.w;
    x[j] = v;
    sum += v.x + v.y + v.z + v.w;
    sq  += v.x * v.x + v.y * v.y + v.z * v.z + v.w * v.w;
  }
#pragma unroll
  for (int o = 16; o > 0; o >>= 1) { sum += __shfl_xor(sum, o, 32); sq += __shfl_xor(sq, o, 32); }
  const float mean = sum * (1.0f / Dn);
  const float var  = sq * (1.0f / Dn) - mean * mean;
  const float inv  = rsqrtf(var + 1e-5f);
#pragma unroll
  for (int j = 0; j < 4; ++j) {
    const int c = (lane + 32 * j) * 4;
    float y0 = (x[j].x - mean) * inv * g[c + 0] + be[c + 0];
    float y1 = (x[j].y - mean) * inv * g[c + 1] + be[c + 1];
    float y2 = (x[j].z - mean) * inv * g[c + 2] + be[c + 2];
    float y3 = (x[j].w - mean) * inv * g[c + 3] + be[c + 3];
    if (outF) {
      float4 o4; o4.x = y0; o4.y = y1; o4.z = y2; o4.w = y3;
      ((float4*)(outF + (size_t)row * Dn))[lane + 32 * j] = o4;
    }
    if (outB) {
      unsigned short* ob = outB + (size_t)row * Dn + c;
      ob[0] = f2bf(y0); ob[1] = f2bf(y1); ob[2] = f2bf(y2); ob[3] = f2bf(y3);
    }
  }
}

// ---------------------------------------------------------------- host side
static inline void launch_gemm(int wnt,
                               const unsigned short* A, long long sA, int lda,
                               const unsigned short* Bm, long long sB, int ldb,
                               float* Cf, unsigned short* Cb,
                               int zdiv, long long csO, long long csI, int ldc,
                               const float* bias, int relu, int M, int N, int K, int batch,
                               hipStream_t stream) {
  dim3 g(N / (64 * wnt), M / 64, batch);
  if (wnt == 2)
    gemm_bf16_wmma_kernel<2><<<g, dim3(256), 0, stream>>>(A, sA, lda, Bm, sB, ldb, Cf, Cb,
                                                          zdiv, csO, csI, ldc, bias, relu, M, N, K);
  else
    gemm_bf16_wmma_kernel<1><<<g, dim3(256), 0, stream>>>(A, sA, lda, Bm, sB, ldb, Cf, Cb,
                                                          zdiv, csO, csI, ldc, bias, relu, M, N, K);
}

extern "C" void kernel_launch(void* const* d_in, const int* in_sizes, int n_in,
                              void* d_out, int out_size, void* d_ws, size_t ws_size,
                              hipStream_t stream) {
  const float* src   = (const float*)d_in[0];
  const float* pf    = (const float*)d_in[1];
  const float* wqkv  = (const float*)d_in[2];
  const float* bqkv  = (const float*)d_in[3];
  const float* wout  = (const float*)d_in[4];
  const float* bout  = (const float*)d_in[5];
  const float* bu    = (const float*)d_in[6];
  const float* bv    = (const float*)d_in[7];
  const float* w1    = (const float*)d_in[8];
  const float* b1    = (const float*)d_in[9];
  const float* w2    = (const float*)d_in[10];
  const float* b2    = (const float*)d_in[11];
  const float* ln1g  = (const float*)d_in[12];
  const float* ln1b  = (const float*)d_in[13];
  const float* ln2g  = (const float*)d_in[14];
  const float* ln2b  = (const float*)d_in[15];
  float* out = (float*)d_out;

  char* wp = (char*)d_ws;
  auto alloc = [&](size_t bytes) -> char* {
    char* p = wp;
    wp += (bytes + 255) & ~(size_t)255;
    return p;
  };
  unsigned short* srcB  = (unsigned short*)alloc((size_t)BLn * Dn * 2);
  unsigned short* wqkvB = (unsigned short*)alloc((size_t)3 * Dn * Dn * 2);
  unsigned short* woutB = (unsigned short*)alloc((size_t)Dn * Dn * 2);
  unsigned short* w1B   = (unsigned short*)alloc((size_t)DFFn * Dn * 2);
  unsigned short* w2B   = (unsigned short*)alloc((size_t)Dn * DFFn * 2);
  float*          qkvF  = (float*)alloc((size_t)BLn * 3 * Dn * 4);
  unsigned short* quB   = (unsigned short*)alloc((size_t)Bn * Hn * Ln * HDn * 2);
  float*          qvF   = (float*)alloc((size_t)Bn * Hn * Ln * HDn * 4);
  unsigned short* kB    = (unsigned short*)alloc((size_t)Bn * Hn * Ln * HDn * 2);
  unsigned short* vTB   = (unsigned short*)alloc((size_t)Bn * Hn * HDn * Ln * 2);
  float*          Sf    = (float*)alloc((size_t)Bn * Hn * Ln * Ln * 4);
  unsigned short* Pb    = (unsigned short*)alloc((size_t)Bn * Hn * Ln * Ln * 2);
  unsigned short* X2b   = (unsigned short*)alloc((size_t)BLn * Dn * 2);
  float*          saF   = (float*)alloc((size_t)BLn * Dn * 4);
  float*          x1F   = (float*)alloc((size_t)BLn * Dn * 4);
  unsigned short* x1B   = (unsigned short*)alloc((size_t)BLn * Dn * 2);
  unsigned short* hB    = (unsigned short*)alloc((size_t)BLn * DFFn * 2);
  float*          ffF   = (float*)alloc((size_t)BLn * Dn * 4);

  auto cvt = [&](const float* in, unsigned short* o, int n) {
    convert_bf16_kernel<<<dim3((n + 255) / 256), dim3(256), 0, stream>>>(in, o, n);
  };

  // 1) fp32 -> bf16 operands
  cvt(src, srcB, BLn * Dn);
  cvt(wqkv, wqkvB, 3 * Dn * Dn);
  cvt(wout, woutB, Dn * Dn);
  cvt(w1, w1B, DFFn * Dn);
  cvt(w2, w2B, Dn * DFFn);

  // 2) QKV projection: qkv = src @ in_proj_w^T + b   (1024 x 1536 x 512)
  launch_gemm(2, srcB, 0, Dn, wqkvB, 0, Dn, qkvF, nullptr, 1, 0, 0, 3 * Dn, bqkv, 0,
              BLn, 3 * Dn, Dn, 1, stream);

  // 3) per-head layouts, q scaling, +bias_u / +bias_v
  prep_qkv_kernel<<<dim3(BLn), dim3(256), 0, stream>>>(qkvF, bu, bv, quB, qvF, kB, vTB);

  // 4) content scores: S[b,h] = (q/s + bu) @ k^T   (32 batches of 256x256x64)
  launch_gemm(2, quB, (long long)Ln * HDn, HDn, kB, (long long)Ln * HDn, HDn, Sf, nullptr,
              1, (long long)Ln * Ln, 0, Ln, nullptr, 0, Ln, Ln, HDn, Bn * Hn, stream);

  // 5) += pairwise term (streams the 537 MB pf tensor once; HBM-bound)
  bd_add_kernel<<<dim3(Ln, Bn), dim3(256), 0, stream>>>(qvF, pf, Sf);

  // 6) softmax -> bf16 probs
  softmax_kernel<<<dim3(Bn * Hn * Ln / 8), dim3(256), 0, stream>>>(Sf, Pb);

  // 7) O = P @ V (N=64 -> WNT=1), scattered bf16 directly into (B,L,D) concat layout
  launch_gemm(1, Pb, (long long)Ln * Ln, Ln, vTB, (long long)HDn * Ln, Ln, nullptr, X2b,
              Hn, (long long)Ln * Dn, (long long)HDn, Dn, nullptr, 0, Ln, HDn, Ln, Bn * Hn, stream);

  // 8) output projection: sa = O @ out_w^T + out_b
  launch_gemm(2, X2b, 0, Dn, woutB, 0, Dn, saF, nullptr, 1, 0, 0, Dn, bout, 0,
              BLn, Dn, Dn, 1, stream);

  // 9) x1 = LN(src + sa)
  add_ln_kernel<<<dim3(BLn / 8), dim3(256), 0, stream>>>(src, saF, ln1g, ln1b, x1F, x1B);

  // 10) FFN1: h = relu(x1 @ lin1_w^T + b1) -> bf16
  launch_gemm(2, x1B, 0, Dn, w1B, 0, Dn, nullptr, hB, 1, 0, 0, DFFn, b1, 1,
              BLn, DFFn, Dn, 1, stream);

  // 11) FFN2: ff = h @ lin2_w^T + b2
  launch_gemm(2, hB, 0, DFFn, w2B, 0, DFFn, ffF, nullptr, 1, 0, 0, Dn, b2, 0,
              BLn, Dn, DFFn, 1, stream);

  // 12) out = LN(x1 + ff)
  add_ln_kernel<<<dim3(BLn / 8), dim3(256), 0, stream>>>(x1F, ffF, ln2g, ln2b, out, nullptr);
}
